// PointNetPlusPlus_3229815407231
// MI455X (gfx1250) — matrix operand order repaired and checked by reading, not compiled, not run on hardware
//
#include <hip/hip_runtime.h>

// ---------------------------------------------------------------------------
// PointNet++ MSG forward for MI455X (gfx1250, wave32, WMMA).
// Pipeline: FPS -> ball query -> fused (gather + 3-layer MLP in LDS via
// v_wmma_f32_16x16x32_f16 + maxpool) per set-abstraction branch.
// BatchNorm folded into weights/bias on device; weights stored f16,
// K-pair interleaved [Kpad/2][Cp][2] so B-fragments load as one b32/lane/VGPR.
// M-register-blocked: each wave holds MT accumulators and reuses one B
// fragment across all M tiles (MT back-to-back v_wmma per B load).
// ---------------------------------------------------------------------------

typedef __attribute__((ext_vector_type(16))) _Float16 v16h;
typedef __attribute__((ext_vector_type(8)))  float    v8f;

#define BATCH 8
#define N0    4096
#define S1    512
#define S2    128
#define EPSBN 1e-5f

// ------------------------------ FPS ---------------------------------------
__global__ __launch_bounds__(256) void fps_kernel(const float* __restrict__ xyz,
                                                  int Npts, int npoint,
                                                  int* __restrict__ outidx) {
  __shared__ float dist[4096];
  __shared__ float redv[256];
  __shared__ int   redi[256];
  __shared__ int   sFar;
  const int b = blockIdx.x, tid = threadIdx.x;
  const float* base = xyz + (size_t)b * Npts * 3;
  for (int i = tid; i < Npts; i += 256) dist[i] = 1e10f;
  if (tid == 0) sFar = 0;
  __syncthreads();
  for (int s = 0; s < npoint; s++) {
    const int f = sFar;
    if (tid == 0) outidx[b * npoint + s] = f;
    const float cx = base[f*3+0], cy = base[f*3+1], cz = base[f*3+2];
    float best = -1.0f; int bi = 0;
    for (int i = tid; i < Npts; i += 256) {
      const float dx = base[i*3+0]-cx, dy = base[i*3+1]-cy, dz = base[i*3+2]-cz;
      const float d = dx*dx + dy*dy + dz*dz;
      const float nd = fminf(dist[i], d);
      dist[i] = nd;
      if (nd > best) { best = nd; bi = i; }
    }
    redv[tid] = best; redi[tid] = bi;
    __syncthreads();
    for (int off = 128; off > 0; off >>= 1) {
      if (tid < off) {
        const float v2 = redv[tid+off]; const int i2 = redi[tid+off];
        if (v2 > redv[tid] || (v2 == redv[tid] && i2 < redi[tid])) {
          redv[tid] = v2; redi[tid] = i2;
        }
      }
      __syncthreads();
    }
    if (tid == 0) sFar = redi[0];
    __syncthreads();
  }
}

// --------------------------- gather sampled xyz ----------------------------
__global__ __launch_bounds__(256) void gather_xyz(const float* __restrict__ xyz,
                                                  const int* __restrict__ idx,
                                                  float* __restrict__ out,
                                                  int S, int Npts) {
  const int t = blockIdx.x * blockDim.x + threadIdx.x;
  if (t >= BATCH * S) return;
  const int b = t / S;
  const int i = idx[t];
  const float* p = xyz + ((size_t)b * Npts + i) * 3;
  out[t*3+0] = p[0]; out[t*3+1] = p[1]; out[t*3+2] = p[2];
}

// ------------------------------ ball query ---------------------------------
// First k indices (ascending) within radius; pad with first found.
__global__ __launch_bounds__(128) void ball_query(const float* __restrict__ xyz,
                                                  const float* __restrict__ new_xyz,
                                                  int Npts, int S, int k, float r2,
                                                  int* __restrict__ gidx) {
  const int t = blockIdx.x * blockDim.x + threadIdx.x;
  if (t >= BATCH * S) return;
  const int b = t / S;
  const float cx = new_xyz[t*3+0], cy = new_xyz[t*3+1], cz = new_xyz[t*3+2];
  const float* base = xyz + (size_t)b * Npts * 3;
  int* out = gidx + (size_t)t * k;
  int cnt = 0, first = 0;
  for (int i = 0; i < Npts && cnt < k; i++) {
    const float dx = base[i*3+0]-cx, dy = base[i*3+1]-cy, dz = base[i*3+2]-cz;
    if (dx*dx + dy*dy + dz*dz <= r2) {
      if (cnt == 0) first = i;
      out[cnt++] = i;
    }
  }
  for (int j = cnt; j < k; j++) out[j] = first;
}

// --------------------------- fold BN into weights --------------------------
// Wp layout: pair-interleaved [Kpad/2][Cp] of u32 = {w[2k][n], w[2k+1][n]} f16.
__global__ __launch_bounds__(256) void fold_weights(const float* __restrict__ W,
                                                    const float* __restrict__ bb,
                                                    const float* __restrict__ g,
                                                    const float* __restrict__ beta,
                                                    const float* __restrict__ rm,
                                                    const float* __restrict__ rv,
                                                    int Cin, int Cout, int Kpad, int Cp,
                                                    unsigned* __restrict__ Wp,
                                                    float* __restrict__ biasOut) {
  const int total = (Kpad / 2) * Cp;
  const int t = blockIdx.x * blockDim.x + threadIdx.x;
  if (t < total) {
    const int k2 = t / Cp, n = t % Cp;
    const float a = (n < Cout) ? g[n] * rsqrtf(rv[n] + EPSBN) : 0.0f;
    const int k0 = 2 * k2, k1 = k0 + 1;
    const float w0 = (n < Cout && k0 < Cin) ? W[(size_t)n * Cin + k0] * a : 0.0f;
    const float w1 = (n < Cout && k1 < Cin) ? W[(size_t)n * Cin + k1] * a : 0.0f;
    union { _Float16 h[2]; unsigned u; } pk;
    pk.h[0] = (_Float16)w0; pk.h[1] = (_Float16)w1;
    Wp[t] = pk.u;
  } else if (t < total + Cp) {
    const int n = t - total;
    float v = 0.0f;
    if (n < Cout) {
      const float a = g[n] * rsqrtf(rv[n] + EPSBN);
      v = (bb[n] - rm[n]) * a + beta[n];
    }
    biasOut[n] = v;
  }
}

// ----------------------- fused SA branch (WMMA core) -----------------------
union AFrag { v16h h; unsigned u[8]; };

// One MLP layer on LDS data, M-register-blocked:
// Y[MT*16 x Cp] = relu(X[MT*16 x Kpad] @ W + bias).
// Each wave owns N-tiles (stride 4) and keeps MT accumulators; one B fragment
// per (nt,kc) feeds MT back-to-back WMMAs. If finalMax, fold rows via LDS
// int-max (all values >= 0 so int compare is order-correct).
template <int MT>
__device__ __forceinline__ void layer_gemm_t(const _Float16* __restrict__ Xin, int Kpad,
                                             const unsigned* __restrict__ Wp,
                                             const float* __restrict__ bias,
                                             int Cp, _Float16* __restrict__ Yout,
                                             int finalMax, int* __restrict__ maxb) {
  const int tid  = threadIdx.x;
  const int wave = tid >> 5;
  const int lane = tid & 31;
  const int nhalf = lane & 15;      // N (B/C/D) or M (A) within tile
  const int grp   = lane >> 4;      // lane half
  const int NT = Cp >> 4;
  const v8f vzero = {0.f, 0.f, 0.f, 0.f, 0.f, 0.f, 0.f, 0.f};
  for (int nt = wave; nt < NT; nt += 4) {
    const int ncol = nt * 16 + nhalf;
    v8f acc[MT];
#pragma unroll
    for (int m = 0; m < MT; m++) acc[m] = vzero;
    for (int kc = 0; kc < Kpad; kc += 32) {
      // B 32x16 f16: VGPR v pair K = grp*16 + 2v  (pair-interleaved storage)
      AFrag bf;
      const int kbase2 = (kc >> 1) + grp * 8;
#pragma unroll
      for (int v = 0; v < 8; v++)
        bf.u[v] = Wp[(size_t)(kbase2 + v) * Cp + ncol];
#pragma unroll
      for (int m = 0; m < MT; m++) {
        // A 16x32 f16: VGPR v pair K = grp*16 + 2v
        AFrag a;
        const unsigned* xrow =
            (const unsigned*)(Xin + (size_t)(m * 16 + nhalf) * Kpad + kc);
#pragma unroll
        for (int v = 0; v < 8; v++) {
          const int kp = grp * 16 + v * 2;
          a.u[v] = xrow[kp >> 1];
        }
        acc[m] = __builtin_amdgcn_wmma_f32_16x16x32_f16(false, a.h, false, bf.h,
                                                        (short)0, acc[m], false, false);
      }
    }
    const float bn = bias[ncol];
    if (finalMax) {
      float mx = 0.0f;
#pragma unroll
      for (int m = 0; m < MT; m++) {
#pragma unroll
        for (int v = 0; v < 8; v++) {
          float y = acc[m][v] + bn;
          y = y > 0.0f ? y : 0.0f;
          mx = y > mx ? y : mx;
        }
      }
      atomicMax(&maxb[ncol], __float_as_int(mx));
    } else {
#pragma unroll
      for (int m = 0; m < MT; m++) {
#pragma unroll
        for (int v = 0; v < 8; v++) {
          float y = acc[m][v] + bn;
          y = y > 0.0f ? y : 0.0f;
          const int mrow = m * 16 + grp * 8 + v;  // D: VGPR v -> row v + 8*grp
          Yout[(size_t)mrow * Cp + ncol] = (_Float16)y;
        }
      }
    }
  }
}

template <int MT>
__device__ __forceinline__ void mlp3(_Float16* X, _Float16* Y, int* maxb,
                                     int Kp1,
                                     const unsigned* W1, const float* b1, int Cp1,
                                     const unsigned* W2, const float* b2, int Cp2,
                                     const unsigned* W3, const float* b3, int Cp3) {
  layer_gemm_t<MT>(X, Kp1, W1, b1, Cp1, Y, 0, nullptr);   // X -> Y
  __syncthreads();
  layer_gemm_t<MT>(Y, Cp1, W2, b2, Cp2, X, 0, nullptr);   // Y -> X
  __syncthreads();
  layer_gemm_t<MT>(X, Cp2, W3, b3, Cp3, nullptr, 1, maxb);// X -> maxpool
  __syncthreads();
}

// mode 1: SA1 (features = [abs xyz, rel xyz], Cin=6)
// mode 2: SA2 (features = [l1feat(320), rel xyz], Cin=323)
// mode 3: SA3 (one block per batch; rows = all S2 points; [xyz, l2feat(512)])
__global__ __launch_bounds__(128) void fused_sa(
    int mode, int S, int Npts, int rows,
    const int* __restrict__ gidx,
    const float* __restrict__ src_xyz,
    const float* __restrict__ new_xyz,
    const _Float16* __restrict__ src_feat, int feat_ch,
    const unsigned* __restrict__ W1, const float* __restrict__ bias1, int Kp1, int Cp1,
    const unsigned* __restrict__ W2, const float* __restrict__ bias2, int Cp2,
    const unsigned* __restrict__ W3, const float* __restrict__ bias3, int Cp3,
    int xbytes, int ybytes,
    _Float16* __restrict__ out_feat, int out_stride, int out_choff,
    float* __restrict__ out_final) {
  extern __shared__ char smem[];
  _Float16* X = (_Float16*)smem;
  _Float16* Y = (_Float16*)(smem + xbytes);
  int* maxb = (int*)(smem + xbytes + ybytes);

  const int blk = blockIdx.x;
  const int b = (mode == 3) ? blk : blk / S;
  const int s = (mode == 3) ? 0 : blk % S;
  const int tid = threadIdx.x;

  for (int n = tid; n < Cp3; n += 128) maxb[n] = 0;

  float cx = 0.f, cy = 0.f, cz = 0.f;
  if (mode != 3) {
    const float* c = new_xyz + ((size_t)b * S + s) * 3;
    cx = c[0]; cy = c[1]; cz = c[2];
  }
  const int tot = rows * Kp1;
  for (int e = tid; e < tot; e += 128) {
    const int r = e / Kp1, c = e % Kp1;
    float val = 0.0f;
    if (mode == 1) {
      const int idx = gidx[((size_t)b * S + s) * rows + r];
      const float* p = src_xyz + ((size_t)b * Npts + idx) * 3;
      if (c < 3)      val = p[c];
      else if (c < 6) val = p[c-3] - (c == 3 ? cx : (c == 4 ? cy : cz));
    } else if (mode == 2) {
      const int idx = gidx[((size_t)b * S + s) * rows + r];
      if (c < feat_ch) {
        val = (float)src_feat[((size_t)b * Npts + idx) * feat_ch + c];
      } else if (c < feat_ch + 3) {
        const int d = c - feat_ch;
        const float* p = src_xyz + ((size_t)b * Npts + idx) * 3;
        val = p[d] - (d == 0 ? cx : (d == 1 ? cy : cz));
      }
    } else {
      if (c < 3)                val = src_xyz[((size_t)b * Npts + r) * 3 + c];
      else if (c < 3 + feat_ch) val = (float)src_feat[((size_t)b * Npts + r) * feat_ch + (c - 3)];
    }
    X[(size_t)r * Kp1 + c] = (_Float16)val;
  }
  __syncthreads();

  if (rows == 32) {
    mlp3<2>(X, Y, maxb, Kp1, W1, bias1, Cp1, W2, bias2, Cp2, W3, bias3, Cp3);
  } else if (rows == 64) {
    mlp3<4>(X, Y, maxb, Kp1, W1, bias1, Cp1, W2, bias2, Cp2, W3, bias3, Cp3);
  } else {
    mlp3<8>(X, Y, maxb, Kp1, W1, bias1, Cp1, W2, bias2, Cp2, W3, bias3, Cp3);
  }

  if (mode == 3) {
    for (int n = tid; n < Cp3; n += 128)
      out_final[(size_t)b * Cp3 + n] = __int_as_float(maxb[n]);
  } else {
    for (int n = tid; n < Cp3; n += 128)
      out_feat[((size_t)b * S + s) * out_stride + out_choff + n] =
          (_Float16)__int_as_float(maxb[n]);
  }
}

// ------------------------------- host side ---------------------------------
static inline int ceil32(int x) { return (x + 31) & ~31; }

extern "C" void kernel_launch(void* const* d_in, const int* in_sizes, int n_in,
                              void* d_out, int out_size, void* d_ws, size_t ws_size,
                              hipStream_t stream) {
  (void)in_sizes; (void)n_in; (void)out_size; (void)ws_size;
  char* ws = (char*)d_ws;
  size_t cur = 0;
  auto alloc = [&](size_t bytes) -> size_t {
    size_t off = (cur + 255) & ~(size_t)255;
    cur = off + bytes;
    return off;
  };

  // 18 layers in d_in order: sa1 b0..b2, sa2 b0..b1, sa3; each layer W,b,g,beta,rm,rv
  static const int dims[18][2] = {
      {6,32},{32,32},{32,64},
      {6,64},{64,64},{64,128},
      {6,64},{64,96},{96,128},
      {323,128},{128,128},{128,256},
      {323,128},{128,196},{196,256},
      {515,256},{256,512},{512,1024}};
  struct LD { int Cin, Cout, Kpad, Cp; size_t wOff, bOff; } L[18];
  for (int i = 0; i < 18; i++) {
    L[i].Cin = dims[i][0]; L[i].Cout = dims[i][1];
    L[i].Kpad = ceil32(L[i].Cin); L[i].Cp = ceil32(L[i].Cout);
    L[i].wOff = alloc((size_t)(L[i].Kpad / 2) * L[i].Cp * 4);
    L[i].bOff = alloc((size_t)L[i].Cp * 4);
  }
  const size_t fps1Off  = alloc((size_t)BATCH * S1 * 4);
  const size_t l1xyzOff = alloc((size_t)BATCH * S1 * 3 * 4);
  const size_t gidx1Off[3] = { alloc((size_t)BATCH*S1*32*4),
                               alloc((size_t)BATCH*S1*64*4),
                               alloc((size_t)BATCH*S1*128*4) };
  const size_t l1featOff = alloc((size_t)BATCH * S1 * 320 * 2);
  const size_t fps2Off   = alloc((size_t)BATCH * S2 * 4);
  const size_t l2xyzOff  = alloc((size_t)BATCH * S2 * 3 * 4);
  const size_t gidx2Off[2] = { alloc((size_t)BATCH*S2*64*4),
                               alloc((size_t)BATCH*S2*128*4) };
  const size_t l2featOff = alloc((size_t)BATCH * S2 * 512 * 2);

  const float* xyz = (const float*)d_in[0];

  // Fold BN into f16 weights.
  for (int i = 0; i < 18; i++) {
    const int p = 1 + i * 6;
    const int total = (L[i].Kpad / 2) * L[i].Cp + L[i].Cp;
    fold_weights<<<(total + 255) / 256, 256, 0, stream>>>(
        (const float*)d_in[p+0], (const float*)d_in[p+1], (const float*)d_in[p+2],
        (const float*)d_in[p+3], (const float*)d_in[p+4], (const float*)d_in[p+5],
        L[i].Cin, L[i].Cout, L[i].Kpad, L[i].Cp,
        (unsigned*)(ws + L[i].wOff), (float*)(ws + L[i].bOff));
  }

  // Allow large dynamic LDS for the fused kernel (SA3 needs ~205 KB).
  (void)hipFuncSetAttribute(reinterpret_cast<const void*>(fused_sa),
                            hipFuncAttributeMaxDynamicSharedMemorySize, 212992);

  // ---- SA1 ----
  fps_kernel<<<BATCH, 256, 0, stream>>>(xyz, N0, S1, (int*)(ws + fps1Off));
  gather_xyz<<<(BATCH*S1 + 255)/256, 256, 0, stream>>>(xyz, (const int*)(ws + fps1Off),
                                                       (float*)(ws + l1xyzOff), S1, N0);
  {
    const int ks[3] = {32, 64, 128};
    const float r2s[3] = {0.01f, 0.04f, 0.16f};
    const int choff[3] = {0, 64, 192};
    for (int br = 0; br < 3; br++) {
      const LD &A = L[br*3+0], &Bl = L[br*3+1], &C = L[br*3+2];
      ball_query<<<(BATCH*S1 + 127)/128, 128, 0, stream>>>(
          xyz, (const float*)(ws + l1xyzOff), N0, S1, ks[br], r2s[br],
          (int*)(ws + gidx1Off[br]));
      const int rows = ks[br];
      const int xb = rows * 2 * (A.Kpad > Bl.Cp ? A.Kpad : Bl.Cp);
      const int yb = rows * 2 * A.Cp;
      const int shmem = xb + yb + C.Cp * 4;
      fused_sa<<<BATCH * S1, 128, shmem, stream>>>(
          1, S1, N0, rows,
          (const int*)(ws + gidx1Off[br]), xyz, (const float*)(ws + l1xyzOff),
          nullptr, 0,
          (const unsigned*)(ws + A.wOff),  (const float*)(ws + A.bOff),  A.Kpad, A.Cp,
          (const unsigned*)(ws + Bl.wOff), (const float*)(ws + Bl.bOff), Bl.Cp,
          (const unsigned*)(ws + C.wOff),  (const float*)(ws + C.bOff),  C.Cp,
          xb, yb,
          (_Float16*)(ws + l1featOff), 320, choff[br], nullptr);
    }
  }

  // ---- SA2 ----
  fps_kernel<<<BATCH, 256, 0, stream>>>((const float*)(ws + l1xyzOff), S1, S2,
                                        (int*)(ws + fps2Off));
  gather_xyz<<<(BATCH*S2 + 255)/256, 256, 0, stream>>>(
      (const float*)(ws + l1xyzOff), (const int*)(ws + fps2Off),
      (float*)(ws + l2xyzOff), S2, S1);
  {
    const int ks[2] = {64, 128};
    const float r2s[2] = {0.16f, 0.64f};
    const int choff[2] = {0, 256};
    for (int br = 0; br < 2; br++) {
      const LD &A = L[9 + br*3 + 0], &Bl = L[9 + br*3 + 1], &C = L[9 + br*3 + 2];
      ball_query<<<(BATCH*S2 + 127)/128, 128, 0, stream>>>(
          (const float*)(ws + l1xyzOff), (const float*)(ws + l2xyzOff),
          S1, S2, ks[br], r2s[br], (int*)(ws + gidx2Off[br]));
      const int rows = ks[br];
      const int xb = rows * 2 * (A.Kpad > Bl.Cp ? A.Kpad : Bl.Cp);
      const int yb = rows * 2 * A.Cp;
      const int shmem = xb + yb + C.Cp * 4;
      fused_sa<<<BATCH * S2, 128, shmem, stream>>>(
          2, S2, S1, rows,
          (const int*)(ws + gidx2Off[br]), (const float*)(ws + l1xyzOff),
          (const float*)(ws + l2xyzOff),
          (const _Float16*)(ws + l1featOff), 320,
          (const unsigned*)(ws + A.wOff),  (const float*)(ws + A.bOff),  A.Kpad, A.Cp,
          (const unsigned*)(ws + Bl.wOff), (const float*)(ws + Bl.bOff), Bl.Cp,
          (const unsigned*)(ws + C.wOff),  (const float*)(ws + C.bOff),  C.Cp,
          xb, yb,
          (_Float16*)(ws + l2featOff), 512, choff[br], nullptr);
    }
  }

  // ---- SA3 (global, one block per batch) ----
  {
    const LD &A = L[15], &Bl = L[16], &C = L[17];
    const int rows = S2;  // 128 points
    const int xb = rows * 2 * (A.Kpad > Bl.Cp ? A.Kpad : Bl.Cp);  // 544 wide
    const int yb = rows * 2 * A.Cp;
    const int shmem = xb + yb + C.Cp * 4;
    fused_sa<<<BATCH, 128, shmem, stream>>>(
        3, 1, S2, rows,
        nullptr, (const float*)(ws + l2xyzOff), nullptr,
        (const _Float16*)(ws + l2featOff), 512,
        (const unsigned*)(ws + A.wOff),  (const float*)(ws + A.bOff),  A.Kpad, A.Cp,
        (const unsigned*)(ws + Bl.wOff), (const float*)(ws + Bl.bOff), Bl.Cp,
        (const unsigned*)(ws + C.wOff),  (const float*)(ws + C.bOff),  C.Cp,
        xb, yb,
        nullptr, 0, 0, (float*)d_out);
  }
}
